// YoloTransformerAdapter_19207093748626
// MI455X (gfx1250) — matrix operand order repaired and checked by reading, not compiled, not run on hardware
//
#include <hip/hip_runtime.h>

typedef __attribute__((ext_vector_type(16))) __bf16 v16bf;
typedef __attribute__((ext_vector_type(8)))  float  v8f;

#define BATCH 8
#define DIM   256
#define HEADS 4
#define DH    64
#define NLAT  256
#define QKSCALE 0.125f   // 64^-0.5

// ---------------------------------------------------------------------------
// WMMA fragment loaders (bf16, 16x16x32), contiguous k order, same in A and B
// A (16x32, row-major [M,K]): lane%16 = row M; lane/16 selects K-halves:
//   elems j -> K = k0 + 16*(lane/16) + j   (contiguous per lane)
__device__ __forceinline__ v16bf load_a_frag(const __bf16* A, int lda, int row,
                                             int k0, int lane) {
  int hi = lane >> 4;
  const __bf16* p = A + (long long)row * lda + k0 + 16 * hi;
  v16bf r;
#pragma unroll
  for (int j = 0; j < 16; ++j) r[j] = p[j];
  return r;
}

// B (32x16) with Bt stored as [N][K] row-major: lane%16 = col N,
// elems j -> K = k0 + 16*(lane/16) + j   (contiguous per lane)
__device__ __forceinline__ v16bf load_bt_frag(const __bf16* Bt, int ldbt,
                                              int col, int k0, int lane) {
  int hi = lane >> 4;
  const __bf16* p = Bt + (long long)col * ldbt + k0 + 16 * hi;
  v16bf r;
#pragma unroll
  for (int j = 0; j < 16; ++j) r[j] = p[j];
  return r;
}

// B stored [K][N] row-major: strided gather
__device__ __forceinline__ v16bf load_b_frag(const __bf16* B, int ldb,
                                             int col, int k0, int lane) {
  int hi = lane >> 4;
  const __bf16* p = B + (long long)(k0 + 16 * hi) * ldb + col;
  v16bf r;
#pragma unroll
  for (int j = 0; j < 16; ++j) r[j] = p[(long long)j * ldb];
  return r;
}

// ---------------------------------------------------------------------------
// Generic GEMM: one wave computes a 16 x (16*NT) C tile, reusing the A
// fragment across NT column subtiles (NT=4 when N%64==0, else NT=1).
// z-batching: z -> (b = z/heads, h = z%heads) with independent stride pairs.
// Epilogue: optional bias[N], exact GELU, f32 residual (flat, z=1 use only),
// output f32 or bf16.
template <bool BT, bool OUTBF, int NT>
__global__ void gemm_k(const __bf16* __restrict__ A, int lda, long long aZb, long long aZh,
                       const __bf16* __restrict__ B, int ldb, long long bZb, long long bZh,
                       void* __restrict__ C, int ldc, long long cZb, long long cZh,
                       int K, int heads,
                       const float* __restrict__ bias,
                       const float* __restrict__ res, int gelu) {
  int lane = threadIdx.x;
  int tn = blockIdx.x, tm = blockIdx.y, z = blockIdx.z;
  int b = z / heads, h = z - b * heads;
  const __bf16* Az = A + b * aZb + h * aZh;
  const __bf16* Bz = B + b * bZb + h * bZh;
  int arow = tm * 16 + (lane & 15);
  int bcol0 = tn * (16 * NT) + (lane & 15);
  v8f acc[NT];
#pragma unroll
  for (int t = 0; t < NT; ++t) acc[t] = {0.f, 0.f, 0.f, 0.f, 0.f, 0.f, 0.f, 0.f};
  for (int k0 = 0; k0 < K; k0 += 32) {
    v16bf af = load_a_frag(Az, lda, arow, k0, lane);
#pragma unroll
    for (int t = 0; t < NT; ++t) {
      v16bf bf = BT ? load_bt_frag(Bz, ldb, bcol0 + 16 * t, k0, lane)
                    : load_b_frag(Bz, ldb, bcol0 + 16 * t, k0, lane);
      acc[t] = __builtin_amdgcn_wmma_f32_16x16x32_bf16(false, af, false, bf,
                                                       (short)0, acc[t], false, false);
    }
  }
  int rbase = tm * 16 + (lane >> 4) * 8;
#pragma unroll
  for (int t = 0; t < NT; ++t) {
    int ccol = tn * (16 * NT) + 16 * t + (lane & 15);
    float bv = bias ? bias[ccol] : 0.0f;
#pragma unroll
    for (int j = 0; j < 8; ++j) {
      int r = rbase + j;
      float v = acc[t][j] + bv;
      if (gelu) v = 0.5f * v * (1.0f + erff(v * 0.70710678f));
      if (res) v += res[(long long)r * ldc + ccol];
      long long idx = b * cZb + h * cZh + (long long)r * ldc + ccol;
      if (OUTBF) ((__bf16*)C)[idx] = (__bf16)v;
      else       ((float*)C)[idx]  = v;
    }
  }
}

// Decoder final projection: C = A[B*np,256] @ wpT + bp, scattered into
// d_out as t_out and out = t_out + feats (un-patchify per reference).
// 16x64 tile per wave (pd is always a multiple of 64).
__global__ void gemm_scatter_k(const __bf16* __restrict__ A,
                               const __bf16* __restrict__ Bt,
                               const float* __restrict__ bias,
                               const float* __restrict__ feats,
                               float* __restrict__ out,
                               long long out_off, long long tout_off,
                               int n, int n_pad, int cdim, int Hc, int Wc) {
  int lane = threadIdx.x;
  int tn = blockIdx.x, tm = blockIdx.y;
  int arow = tm * 16 + (lane & 15);
  int bcol0 = tn * 64 + (lane & 15);
  v8f acc[4];
#pragma unroll
  for (int t = 0; t < 4; ++t) acc[t] = {0.f, 0.f, 0.f, 0.f, 0.f, 0.f, 0.f, 0.f};
  for (int k0 = 0; k0 < DIM; k0 += 32) {
    v16bf af = load_a_frag(A, DIM, arow, k0, lane);
#pragma unroll
    for (int t = 0; t < 4; ++t) {
      v16bf bf = load_bt_frag(Bt, DIM, bcol0 + 16 * t, k0, lane);
      acc[t] = __builtin_amdgcn_wmma_f32_16x16x32_bf16(false, af, false, bf,
                                                       (short)0, acc[t], false, false);
    }
  }
  int rbase = tm * 16 + (lane >> 4) * 8;
  int num_w = Hc >> 2;  // reference: num_w = h//P
  long long HW = (long long)Hc * Wc;
#pragma unroll
  for (int t = 0; t < 4; ++t) {
    int jj = tn * 64 + 16 * t + (lane & 15);
    int ch = jj % cdim;
    int tt = jj / cdim;
    int p2 = tt & 3, p1 = tt >> 2;
    float bv = bias[jj];
#pragma unroll
    for (int j = 0; j < 8; ++j) {
      int r = rbase + j;
      int b = r / n_pad;
      int p = r - b * n_pad;
      if (p >= n) continue;
      int ih = p / num_w, iw = p - ih * num_w;
      int s = (ih * 4 + p1) * Hc + iw * 4 + p2;  // flat spatial index
      long long base = ((long long)b * cdim + ch) * HW + s;
      float v = acc[t][j] + bv;
      out[tout_off + base] = v;
      out[out_off + base]  = v + feats[base];
    }
  }
}

// ---------------------------------------------------------------------------
// emb[b,p,j] = feats[b, ch, y0*4+p1, x0*4+p2] + pos[p, j]  (bf16, zero-padded)
__global__ void patch_embed_k(const float* __restrict__ f,
                              const float* __restrict__ pos,
                              __bf16* __restrict__ emb,
                              int n, int n_pad, int pd, int cdim, int Hc, int Wc) {
  long long i = blockIdx.x * (long long)blockDim.x + threadIdx.x;
  long long tot = (long long)BATCH * n_pad * pd;
  if (i >= tot) return;
  int j = (int)(i % pd);
  long long t1 = i / pd;
  int p = (int)(t1 % n_pad);
  int b = (int)(t1 / n_pad);
  __bf16 v;
  if (p >= n) {
    v = (__bf16)0.0f;
  } else {
    int ch = j % cdim;
    int t = j / cdim;
    int p2 = t & 3, p1 = t >> 2;
    int nw = Wc >> 2;
    int y0 = p / nw, x0 = p - y0 * nw;
    float x = f[((long long)b * cdim + ch) * (Hc * Wc) + (y0 * 4 + p1) * Wc + (x0 * 4 + p2)];
    v = (__bf16)(x + pos[(long long)p * pd + j]);
  }
  emb[i] = v;
}

// LayerNorm over rows of length 256 -> bf16; rows >= valid written as zero.
__global__ void ln_k(const float* __restrict__ X, const float* __restrict__ g,
                     const float* __restrict__ bb, __bf16* __restrict__ Y,
                     int rows, int valid) {
  int row = blockIdx.x * blockDim.y + threadIdx.y;
  if (row >= rows) return;
  int lane = threadIdx.x;
  __bf16* yr = Y + (long long)row * DIM + lane * 8;
  if (row >= valid) {
#pragma unroll
    for (int j = 0; j < 8; ++j) yr[j] = (__bf16)0.0f;
    return;
  }
  const float* xr = X + (long long)row * DIM + lane * 8;
  float v[8], s = 0.f, sq = 0.f;
#pragma unroll
  for (int j = 0; j < 8; ++j) { v[j] = xr[j]; s += v[j]; sq += v[j] * v[j]; }
  for (int off = 16; off; off >>= 1) {
    s += __shfl_xor(s, off, 32);
    sq += __shfl_xor(sq, off, 32);
  }
  float m = s * (1.0f / 256.0f);
  float var = sq * (1.0f / 256.0f) - m * m;
  float rs = rsqrtf(var + 1e-5f);
#pragma unroll
  for (int j = 0; j < 8; ++j) {
    int col = lane * 8 + j;
    yr[j] = (__bf16)((v[j] - m) * rs * g[col] + bb[col]);
  }
}

// Stable softmax over rows of length L (valid mvalid), applies QKSCALE, bf16 out.
__global__ void softmax_k(const float* __restrict__ S, __bf16* __restrict__ Aout,
                          int L, int mvalid, int rows) {
  int row = blockIdx.x * blockDim.y + threadIdx.y;
  if (row >= rows) return;
  int lane = threadIdx.x;
  const float* sr = S + (long long)row * L;
  __bf16* ar = Aout + (long long)row * L;
  float mx = -3.4e38f;
  for (int j = lane; j < mvalid; j += 32) mx = fmaxf(mx, sr[j] * QKSCALE);
  for (int off = 16; off; off >>= 1) mx = fmaxf(mx, __shfl_xor(mx, off, 32));
  float sum = 0.f;
  for (int j = lane; j < mvalid; j += 32) sum += __expf(sr[j] * QKSCALE - mx);
  for (int off = 16; off; off >>= 1) sum += __shfl_xor(sum, off, 32);
  float inv = 1.0f / sum;
  for (int j = lane; j < L; j += 32) {
    float v = (j < mvalid) ? __expf(sr[j] * QKSCALE - mx) * inv : 0.0f;
    ar[j] = (__bf16)v;
  }
}

// Wt[N][K] = bf16(W[K][N])
__global__ void wtrans_k(const float* __restrict__ W, __bf16* __restrict__ Wt,
                         int Kr, int Nc) {
  long long i = blockIdx.x * (long long)blockDim.x + threadIdx.x;
  long long tot = (long long)Kr * Nc;
  if (i >= tot) return;
  int nIdx = (int)(i / Kr);
  int kIdx = (int)(i % Kr);
  Wt[i] = (__bf16)W[(long long)kIdx * Nc + nIdx];
}

__global__ void bcast_k(const float* __restrict__ lat, float* __restrict__ x, int total) {
  int i = blockIdx.x * blockDim.x + threadIdx.x;
  if (i < total) x[i] = lat[i & (NLAT * DIM - 1)];
}

// ---------------------------------------------------------------------------
extern "C" void kernel_launch(void* const* d_in, const int* in_sizes, int n_in,
                              void* d_out_v, int out_size, void* d_ws, size_t ws_size,
                              hipStream_t stream) {
  (void)in_sizes; (void)n_in; (void)out_size; (void)ws_size;
  float* d_out = (float*)d_out_v;
  const float* featsP[4] = {(const float*)d_in[0], (const float*)d_in[1],
                            (const float*)d_in[2], (const float*)d_in[3]};
  auto Lf = [&](int i) { return (const float*)d_in[i]; };
  // JAX sorted-key pytree flattening: dec(4x13), enc(4x13), latents, self(2x12)
  // dec leaf: 0 a.bo 1 a.ln_b 2 a.ln_g 3 a.wkv 4 a.wo 5 a.wq 6 bp 7 ctx_ln_b
  //           8 ctx_ln_g 9 ln_b 10 ln_g 11 queries 12 wp
  // enc leaf: 0 a.bo 1 a.ln_b 2 a.ln_g 3 a.wkv 4 a.wo 5 a.wq 6 b1 7 b2
  //           8 f.ln_b 9 f.ln_g 10 w1 11 w2 12 pos
  // self leaf: 0 a.bo 1 a.ln_b 2 a.ln_g 3 a.wkv 4 a.wo 5 a.wq 6 b1 7 b2
  //           8 f.ln_b 9 f.ln_g 10 w1 11 w2
  const int DEC0 = 4, ENC0 = 4 + 52, LAT = 4 + 104, SELF0 = 4 + 105;

  const int Cc[4] = {256, 512, 768, 1024};
  const int Hh[4] = {96, 48, 24, 12};
  const int Ww[4] = {160, 80, 40, 20};
  int nA[4], npA[4], pdA[4];
  long long fsz[4], outoff[4], toutoff[4];
  for (int i = 0; i < 4; ++i) {
    nA[i] = (Hh[i] / 4) * (Ww[i] / 4);
    npA[i] = ((nA[i] + 31) / 32) * 32;
    pdA[i] = Cc[i] * 16;
    fsz[i] = (long long)BATCH * Cc[i] * Hh[i] * Ww[i];
  }
  outoff[0] = 0;
  for (int i = 1; i < 4; ++i) outoff[i] = outoff[i - 1] + fsz[i - 1];
  long long xoff = outoff[3] + fsz[3];
  toutoff[0] = xoff + (long long)BATCH * NLAT * DIM;
  for (int i = 1; i < 4; ++i) toutoff[i] = toutoff[i - 1] + fsz[i - 1];

  // Workspace layout (bytes)
  char* ws = (char*)d_ws;
  const size_t MB = (size_t)1 << 20;
  float*  xA   = (float*)(ws + 0 * MB);    // 2 MB
  float*  xB   = (float*)(ws + 2 * MB);    // 2 MB
  __bf16* xn   = (__bf16*)(ws + 4 * MB);   // 1 MB
  __bf16* qb   = (__bf16*)(ws + 5 * MB);   // 1 MB
  __bf16* kvb  = (__bf16*)(ws + 6 * MB);   // 8 MB
  __bf16* ob   = (__bf16*)(ws + 14 * MB);  // 4 MB
  __bf16* aob  = (__bf16*)(ws + 18 * MB);  // 4 MB
  __bf16* h1b  = (__bf16*)(ws + 22 * MB);  // 1 MB
  __bf16* qnb  = (__bf16*)(ws + 23 * MB);  // 1 MB
  __bf16* wT   = (__bf16*)(ws + 24 * MB);  // 17 MB
  char*   region = ws + 41 * MB;           // 63 MB, emb OR scores+a
  float*  scores = (float*)region;
  __bf16* abuf   = (__bf16*)(region + 32 * MB);
  __bf16* emb    = (__bf16*)region;

  auto ln = [&](const float* X, const float* g, const float* b_, __bf16* Y,
                int rows, int valid) {
    dim3 blk(32, 8);
    ln_k<<<dim3((rows + 7) / 8), blk, 0, stream>>>(X, g, b_, Y, rows, valid);
  };
  auto wtrans = [&](const float* W, __bf16* Wt, int Kr, int Nc) {
    long long tot = (long long)Kr * Nc;
    wtrans_k<<<(unsigned)((tot + 255) / 256), 256, 0, stream>>>(W, Wt, Kr, Nc);
  };
  auto softmax = [&](const float* S, __bf16* Ao, int Lr, int mval, int rows) {
    dim3 blk(32, 4);
    softmax_k<<<dim3((rows + 3) / 4), blk, 0, stream>>>(S, Ao, Lr, mval, rows);
  };
  auto gemm = [&](bool bt, bool outbf,
                  const __bf16* A, int lda, long long aZb, long long aZh,
                  const __bf16* B, int ldb, long long bZb, long long bZh,
                  void* C, int ldc, long long cZb, long long cZh,
                  int M, int N, int K, int Z, int heads,
                  const float* bias, const float* res, int gelu) {
    dim3 blk(32);
    if ((N & 63) == 0) {
      dim3 grd(N / 64, M / 16, Z);
      if (bt) {
        if (outbf) gemm_k<true, true, 4><<<grd, blk, 0, stream>>>(A, lda, aZb, aZh, B, ldb, bZb, bZh, C, ldc, cZb, cZh, K, heads, bias, res, gelu);
        else       gemm_k<true, false, 4><<<grd, blk, 0, stream>>>(A, lda, aZb, aZh, B, ldb, bZb, bZh, C, ldc, cZb, cZh, K, heads, bias, res, gelu);
      } else {
        if (outbf) gemm_k<false, true, 4><<<grd, blk, 0, stream>>>(A, lda, aZb, aZh, B, ldb, bZb, bZh, C, ldc, cZb, cZh, K, heads, bias, res, gelu);
        else       gemm_k<false, false, 4><<<grd, blk, 0, stream>>>(A, lda, aZb, aZh, B, ldb, bZb, bZh, C, ldc, cZb, cZh, K, heads, bias, res, gelu);
      }
    } else {
      dim3 grd(N / 16, M / 16, Z);
      if (bt) {
        if (outbf) gemm_k<true, true, 1><<<grd, blk, 0, stream>>>(A, lda, aZb, aZh, B, ldb, bZb, bZh, C, ldc, cZb, cZh, K, heads, bias, res, gelu);
        else       gemm_k<true, false, 1><<<grd, blk, 0, stream>>>(A, lda, aZb, aZh, B, ldb, bZb, bZh, C, ldc, cZb, cZh, K, heads, bias, res, gelu);
      } else {
        if (outbf) gemm_k<false, true, 1><<<grd, blk, 0, stream>>>(A, lda, aZb, aZh, B, ldb, bZb, bZh, C, ldc, cZb, cZh, K, heads, bias, res, gelu);
        else       gemm_k<false, false, 1><<<grd, blk, 0, stream>>>(A, lda, aZb, aZh, B, ldb, bZb, bZh, C, ldc, cZb, cZh, K, heads, bias, res, gelu);
      }
    }
  };

  // x = broadcast(latents)
  int xtot = BATCH * NLAT * DIM;
  bcast_k<<<(xtot + 255) / 256, 256, 0, stream>>>(Lf(LAT), xA, xtot);
  float* xcur = xA;
  float* xalt = xB;
  const int Mx = BATCH * NLAT;  // 2048

  // ---------------- Encoder: per-scale cross-attn + FF ----------------
  for (int i = 0; i < 4; ++i) {
    const float* bo    = Lf(ENC0 + i * 13 + 0);
    const float* alnb  = Lf(ENC0 + i * 13 + 1);
    const float* alng  = Lf(ENC0 + i * 13 + 2);
    const float* wkv   = Lf(ENC0 + i * 13 + 3);
    const float* wo    = Lf(ENC0 + i * 13 + 4);
    const float* wq    = Lf(ENC0 + i * 13 + 5);
    const float* b1    = Lf(ENC0 + i * 13 + 6);
    const float* b2    = Lf(ENC0 + i * 13 + 7);
    const float* flnb  = Lf(ENC0 + i * 13 + 8);
    const float* flng  = Lf(ENC0 + i * 13 + 9);
    const float* w1    = Lf(ENC0 + i * 13 + 10);
    const float* w2    = Lf(ENC0 + i * 13 + 11);
    const float* pos   = Lf(ENC0 + i * 13 + 12);
    int n = nA[i], np = npA[i], pd = pdA[i];

    ln(xcur, alng, alnb, xn, Mx, Mx);
    wtrans(wq, wT, DIM, DIM);
    gemm(true, true, xn, DIM, 0, 0, wT, DIM, 0, 0, qb, DIM, 0, 0,
         Mx, DIM, DIM, 1, 1, nullptr, nullptr, 0);

    long long tot = (long long)BATCH * np * pd;
    patch_embed_k<<<(unsigned)((tot + 255) / 256), 256, 0, stream>>>(
        featsP[i], pos, emb, n, np, pd, Cc[i], Hh[i], Ww[i]);
    wtrans(wkv, wT, pd, 2 * DIM);
    gemm(true, true, emb, pd, 0, 0, wT, pd, 0, 0, kvb, 2 * DIM, 0, 0,
         BATCH * np, 2 * DIM, pd, 1, 1, nullptr, nullptr, 0);

    // scores[b,h,256,np] = q_h @ k_h^T   (emb region now free -> scores)
    gemm(true, false, qb, DIM, (long long)NLAT * DIM, DH,
         kvb, 2 * DIM, (long long)np * 2 * DIM, DH,
         scores, np, (long long)HEADS * NLAT * np, (long long)NLAT * np,
         NLAT, np, DH, BATCH * HEADS, HEADS, nullptr, nullptr, 0);
    softmax(scores, abuf, np, n, BATCH * HEADS * NLAT);
    // o[b,256, h*64..] = a @ v_h
    gemm(false, true, abuf, np, (long long)HEADS * NLAT * np, (long long)NLAT * np,
         kvb + DIM, 2 * DIM, (long long)np * 2 * DIM, DH,
         ob, DIM, (long long)NLAT * DIM, DH,
         NLAT, DH, np, BATCH * HEADS, HEADS, nullptr, nullptr, 0);
    // x = o @ wo + bo + x
    wtrans(wo, wT, DIM, DIM);
    gemm(true, false, ob, DIM, 0, 0, wT, DIM, 0, 0, xalt, DIM, 0, 0,
         Mx, DIM, DIM, 1, 1, bo, xcur, 0);
    { float* t = xcur; xcur = xalt; xalt = t; }
    // FF
    ln(xcur, flng, flnb, xn, Mx, Mx);
    wtrans(w1, wT, DIM, DIM);
    gemm(true, true, xn, DIM, 0, 0, wT, DIM, 0, 0, h1b, DIM, 0, 0,
         Mx, DIM, DIM, 1, 1, b1, nullptr, 1);
    wtrans(w2, wT, DIM, DIM);
    gemm(true, false, h1b, DIM, 0, 0, wT, DIM, 0, 0, xalt, DIM, 0, 0,
         Mx, DIM, DIM, 1, 1, b2, xcur, 0);
    { float* t = xcur; xcur = xalt; xalt = t; }
  }

  // ---------------- Self-attention trunk ----------------
  for (int l = 0; l < 2; ++l) {
    const float* bo   = Lf(SELF0 + l * 12 + 0);
    const float* alnb = Lf(SELF0 + l * 12 + 1);
    const float* alng = Lf(SELF0 + l * 12 + 2);
    const float* wkv  = Lf(SELF0 + l * 12 + 3);
    const float* wo   = Lf(SELF0 + l * 12 + 4);
    const float* wq   = Lf(SELF0 + l * 12 + 5);
    const float* b1   = Lf(SELF0 + l * 12 + 6);
    const float* b2   = Lf(SELF0 + l * 12 + 7);
    const float* flnb = Lf(SELF0 + l * 12 + 8);
    const float* flng = Lf(SELF0 + l * 12 + 9);
    const float* w1   = Lf(SELF0 + l * 12 + 10);
    const float* w2   = Lf(SELF0 + l * 12 + 11);

    ln(xcur, alng, alnb, xn, Mx, Mx);
    wtrans(wq, wT, DIM, DIM);
    gemm(true, true, xn, DIM, 0, 0, wT, DIM, 0, 0, qb, DIM, 0, 0,
         Mx, DIM, DIM, 1, 1, nullptr, nullptr, 0);
    wtrans(wkv, wT, DIM, 2 * DIM);
    gemm(true, true, xn, DIM, 0, 0, wT, DIM, 0, 0, kvb, 2 * DIM, 0, 0,
         Mx, 2 * DIM, DIM, 1, 1, nullptr, nullptr, 0);
    gemm(true, false, qb, DIM, (long long)NLAT * DIM, DH,
         kvb, 2 * DIM, (long long)NLAT * 2 * DIM, DH,
         scores, NLAT, (long long)HEADS * NLAT * NLAT, (long long)NLAT * NLAT,
         NLAT, NLAT, DH, BATCH * HEADS, HEADS, nullptr, nullptr, 0);
    softmax(scores, abuf, NLAT, NLAT, BATCH * HEADS * NLAT);
    gemm(false, true, abuf, NLAT, (long long)HEADS * NLAT * NLAT, (long long)NLAT * NLAT,
         kvb + DIM, 2 * DIM, (long long)NLAT * 2 * DIM, DH,
         ob, DIM, (long long)NLAT * DIM, DH,
         NLAT, DH, NLAT, BATCH * HEADS, HEADS, nullptr, nullptr, 0);
    wtrans(wo, wT, DIM, DIM);
    gemm(true, false, ob, DIM, 0, 0, wT, DIM, 0, 0, xalt, DIM, 0, 0,
         Mx, DIM, DIM, 1, 1, bo, xcur, 0);
    { float* t = xcur; xcur = xalt; xalt = t; }
    ln(xcur, flng, flnb, xn, Mx, Mx);
    wtrans(w1, wT, DIM, DIM);
    gemm(true, true, xn, DIM, 0, 0, wT, DIM, 0, 0, h1b, DIM, 0, 0,
         Mx, DIM, DIM, 1, 1, b1, nullptr, 1);
    wtrans(w2, wT, DIM, DIM);
    gemm(true, false, h1b, DIM, 0, 0, wT, DIM, 0, 0, xalt, DIM, 0, 0,
         Mx, DIM, DIM, 1, 1, b2, xcur, 0);
    { float* t = xcur; xcur = xalt; xalt = t; }
  }

  // emit latent output x
  hipMemcpyAsync(d_out + xoff, xcur, sizeof(float) * (size_t)Mx * DIM,
                 hipMemcpyDeviceToDevice, stream);

  // ---------------- Decoder: per-scale query cross-attn ----------------
  for (int i = 0; i < 4; ++i) {
    const float* bo      = Lf(DEC0 + i * 13 + 0);
    const float* wkv     = Lf(DEC0 + i * 13 + 3);
    const float* wo      = Lf(DEC0 + i * 13 + 4);
    const float* wq      = Lf(DEC0 + i * 13 + 5);
    const float* bp      = Lf(DEC0 + i * 13 + 6);
    const float* clnb    = Lf(DEC0 + i * 13 + 7);
    const float* clng    = Lf(DEC0 + i * 13 + 8);
    const float* lnb     = Lf(DEC0 + i * 13 + 9);
    const float* lng     = Lf(DEC0 + i * 13 + 10);
    const float* queries = Lf(DEC0 + i * 13 + 11);
    const float* wp      = Lf(DEC0 + i * 13 + 12);
    int n = nA[i], np = npA[i], pd = pdA[i];

    // qn = LN(queries)  (batch-independent); pad rows zeroed
    ln(queries, lng, lnb, qnb, np, n);
    wtrans(wq, wT, DIM, DIM);
    gemm(true, true, qnb, DIM, 0, 0, wT, DIM, 0, 0, qb, DIM, 0, 0,
         np, DIM, DIM, 1, 1, nullptr, nullptr, 0);
    // ctxn = LN(x)
    ln(xcur, clng, clnb, xn, Mx, Mx);
    wtrans(wkv, wT, DIM, 2 * DIM);
    gemm(true, true, xn, DIM, 0, 0, wT, DIM, 0, 0, kvb, 2 * DIM, 0, 0,
         Mx, 2 * DIM, DIM, 1, 1, nullptr, nullptr, 0);
    // scores[b,h,np,256]
    gemm(true, false, qb, DIM, 0, DH,
         kvb, 2 * DIM, (long long)NLAT * 2 * DIM, DH,
         scores, NLAT, (long long)HEADS * np * NLAT, (long long)np * NLAT,
         np, NLAT, DH, BATCH * HEADS, HEADS, nullptr, nullptr, 0);
    softmax(scores, abuf, NLAT, NLAT, BATCH * HEADS * np);
    // o[b,np,256]
    gemm(false, true, abuf, NLAT, (long long)HEADS * np * NLAT, (long long)np * NLAT,
         kvb + DIM, 2 * DIM, (long long)NLAT * 2 * DIM, DH,
         ob, DIM, (long long)np * DIM, DH,
         np, DH, NLAT, BATCH * HEADS, HEADS, nullptr, nullptr, 0);
    // attn_out = o @ wo + bo  (bf16)
    wtrans(wo, wT, DIM, DIM);
    gemm(true, true, ob, DIM, 0, 0, wT, DIM, 0, 0, aob, DIM, 0, 0,
         BATCH * np, DIM, DIM, 1, 1, bo, nullptr, 0);
    // final projection + un-patchify scatter into d_out (t_out and out)
    wtrans(wp, wT, DIM, pd);
    dim3 grd(pd / 64, (BATCH * np) / 16);
    gemm_scatter_k<<<grd, 32, 0, stream>>>(aob, wT, bp, featsP[i], d_out,
                                           outoff[i], toutoff[i],
                                           n, np, Cc[i], Hh[i], Ww[i]);
  }
}
